// add_edges_2241972928588
// MI455X (gfx1250) — matrix operand order repaired and checked by reading, not compiled, not run on hardware
//
#include <hip/hip_runtime.h>
#include <hip/hip_bf16.h>

typedef __attribute__((ext_vector_type(16))) _Float16 v16h;
typedef __attribute__((ext_vector_type(8)))  float    v8f;
typedef __attribute__((ext_vector_type(4)))  int      vi4;

#define SAMPLING_RATE 0.5f
#define TAU           0.3f
#define EPSF          1e-8f

// ---------------------------------------------------------------------------
// helpers
// ---------------------------------------------------------------------------
__device__ __forceinline__ unsigned f2o(float f) {
    // order-preserving float -> uint transform (ascending)
    unsigned u = __float_as_uint(f);
    return (u & 0x80000000u) ? ~u : (u | 0x80000000u);
}

__device__ __forceinline__ float hash_uniform(unsigned idx, unsigned salt) {
    unsigned x = idx * 0x9E3779B9u ^ (salt + 1u) * 0x85EBCA6Bu ^ 42u;
    x ^= x >> 16; x *= 0x7FEB352Du;
    x ^= x >> 15; x *= 0x846CA68Bu;
    x ^= x >> 16;
    return ((float)x + 0.5f) * (1.0f / 4294967296.0f); // (0,1)
}

// ---------------------------------------------------------------------------
// gfx1250 async global->LDS copy (ASYNCcnt path), with safe fallback
// Probe-confirmed: param 0 is  v4i addrspace(1)*  (global src),
// param 1 is the LDS destination, then imm offset + imm cpol.
// ---------------------------------------------------------------------------
#if __has_builtin(__builtin_amdgcn_global_load_async_to_lds_b128)
#define HAS_ASYNC_LDS 1
#else
#define HAS_ASYNC_LDS 0
#endif

__device__ __forceinline__ void async_copy16(void* lds_dst, const void* gsrc) {
#if HAS_ASYNC_LDS
    __builtin_amdgcn_global_load_async_to_lds_b128(
        (__attribute__((address_space(1))) vi4*)gsrc,
        (__attribute__((address_space(3))) vi4*)lds_dst,
        0, 0);
#else
    *(uint4*)lds_dst = *(const uint4*)gsrc;
#endif
}

__device__ __forceinline__ void async_wait_all() {
#if HAS_ASYNC_LDS
#if __has_builtin(__builtin_amdgcn_s_wait_asynccnt)
    __builtin_amdgcn_s_wait_asynccnt(0);
#else
    asm volatile("s_wait_asynccnt 0x0" ::: "memory");
#endif
#endif
}

// ---------------------------------------------------------------------------
// Kernel A: f32 -> f16 feature copy + per-row squared norms
// grid = N rows, block = 128
// ---------------------------------------------------------------------------
__global__ void feat_norm_kernel(const float* __restrict__ feat,
                                 _Float16* __restrict__ feat_h,
                                 float* __restrict__ norms) {
    int row = blockIdx.x;
    int tid = threadIdx.x;
    float v = feat[(size_t)row * 128 + tid];
    feat_h[(size_t)row * 128 + tid] = (_Float16)v;
    __shared__ float s[128];
    s[tid] = v * v;
    __syncthreads();
    for (int o = 64; o > 0; o >>= 1) {
        if (tid < o) s[tid] += s[tid + o];
        __syncthreads();
    }
    if (tid == 0) norms[row] = s[0];
}

// ---------------------------------------------------------------------------
// Kernel B: per-edge T = ||xi - xj|| * e_i * e_j using WMMA Gram-tile diagonal
// One wave (32 threads) handles 16 edges. 4x v_wmma_f32_16x16x32_f16 (K=128).
// Feature rows are gathered into LDS with async global->LDS 128b transfers.
// ---------------------------------------------------------------------------
__global__ void edge_T_wmma_kernel(const int* __restrict__ cand_i,
                                   const int* __restrict__ cand_j,
                                   const _Float16* __restrict__ feat_h,
                                   const float* __restrict__ norms,
                                   const float* __restrict__ energy,
                                   float* __restrict__ T,
                                   int Ec) {
    __shared__ _Float16 sXi[16][128];  // A rows  (x_i)   4KB
    __shared__ _Float16 sXj[16][128];  // B cols  (x_j)   4KB
    __shared__ float    sC[8][32];     // C/D staging for diagonal extraction
    __shared__ int      sNi[16], sNj[16];

    int lane = threadIdx.x;                // 0..31 (wave32)
    int e    = blockIdx.x * 16 + (lane & 15);

    if (lane < 16) {
        int ee = e < Ec ? e : Ec - 1;
        sNi[lane] = cand_i[ee];
        sNj[lane] = cand_j[ee];
    }
    __syncthreads();

    // gather 16 rows of xi and 16 rows of xj into LDS.
    // 32 rows x 256B = 512 chunks of 16B; 32 lanes x 16 iterations, per-lane
    // addressed async DMA (GLOBAL_LOAD_ASYNC_TO_LDS_B128, ASYNCcnt-tracked).
    for (int t = 0; t < 16; ++t) {
        int c   = t * 32 + lane;
        int row = c >> 4;          // 0..31
        int ci  = c & 15;          // 16B chunk within the 256B row
        int node = (row < 16) ? sNi[row] : sNj[row - 16];
        const char* gsrc = (const char*)feat_h + (size_t)node * 256 + ci * 16;
        char* ldst = (row < 16) ? ((char*)&sXi[row][0] + ci * 16)
                                : ((char*)&sXj[row - 16][0] + ci * 16);
        async_copy16(ldst, gsrc);
    }
    async_wait_all();
    __syncthreads();

    // A fragment layout (16-bit A 16x32, ISA 7.12.2):
    //   lanes 0-15 : M=lane,   K = 0..7 (v0..3) and 16..23 (v4..7)
    //   lanes 16-31: M=lane-16,K = 8..15         and 24..31
    // B fragment (32x16): same K order as A; lanes 0-15 -> N=lane, K 0..7,16..23;
    //   lanes 16-31 -> N=lane-16, K 8..15,24..31.
    int rowA = lane & 15;
    int hiA  = lane >> 4;

    v8f c = {};
    for (int kc = 0; kc < 4; ++kc) {
        v16h a, b;
        for (int idx = 0; idx < 16; ++idx) {
            int ka = kc * 32 + ((idx & 8) << 1) + (idx & 7) + (hiA ? 8 : 0);
            a[idx] = sXi[rowA][ka];
            int kb = kc * 32 + ((idx & 8) << 1) + (idx & 7) + (hiA ? 8 : 0);
            b[idx] = sXj[rowA][kb];
        }
        c = __builtin_amdgcn_wmma_f32_16x16x32_f16(
                false, a, false, b, (short)0, c, false, false);
    }

    // extract diagonal of the 16x16 Gram tile: C vgpr r / lane l<16 -> (M=r,N=l);
    // lane l>=16 -> (M=r+8, N=l-16)
    for (int v = 0; v < 8; ++v) sC[v][lane] = c[v];
    __syncthreads();

    if (lane < 16 && e < Ec) {
        int   m   = lane;
        float dot = (m < 8) ? sC[m][m] : sC[m - 8][m + 16];
        int ni = sNi[m], nj = sNj[m];
        float d2 = norms[ni] + norms[nj] - 2.0f * dot;
        float d  = sqrtf(fmaxf(d2, 0.0f));
        T[e] = d * energy[ni] * energy[nj];
    }
}

// ---------------------------------------------------------------------------
// Sort-1 key init: descending-T stable sort == ascending sort of ~f2o(T)
// ---------------------------------------------------------------------------
__global__ void key1_kernel(const float* __restrict__ T,
                            unsigned* __restrict__ keys,
                            unsigned* __restrict__ vals, int n) {
    int t = blockIdx.x * blockDim.x + threadIdx.x;
    if (t < n) { keys[t] = ~f2o(T[t]); vals[t] = (unsigned)t; }
}

// ---------------------------------------------------------------------------
// Radix sort: hist -> scan -> stable ballot scatter.  256 bins, 4096 elem/blk.
// ---------------------------------------------------------------------------
#define RS_ELEMS 4096

__global__ void radix_hist_kernel(const unsigned* __restrict__ keys, int n,
                                  int shift, unsigned* __restrict__ hist,
                                  int numBlocks) {
    __shared__ unsigned h[256];
    int tid = threadIdx.x;
    h[tid] = 0;
    __syncthreads();
    int base = blockIdx.x * RS_ELEMS;
    for (int t = tid; t < RS_ELEMS; t += 256) {
        int idx = base + t;
        if (idx < n) atomicAdd(&h[(keys[idx] >> shift) & 255u], 1u);
    }
    __syncthreads();
    hist[(size_t)tid * numBlocks + blockIdx.x] = h[tid];  // bin-major
}

__global__ void radix_scan_kernel(unsigned* __restrict__ hist, int total) {
    __shared__ unsigned tmp[256];
    __shared__ unsigned carry;
    int tid = threadIdx.x;
    if (tid == 0) carry = 0;
    __syncthreads();
    for (int base = 0; base < total; base += 256) {
        int i = base + tid;
        unsigned v = (i < total) ? hist[i] : 0u;
        tmp[tid] = v;
        __syncthreads();
        for (int off = 1; off < 256; off <<= 1) {
            unsigned add = (tid >= off) ? tmp[tid - off] : 0u;
            __syncthreads();
            tmp[tid] += add;
            __syncthreads();
        }
        unsigned incl = tmp[tid];
        unsigned cbase = carry;
        if (i < total) hist[i] = cbase + incl - v;  // exclusive + carry
        __syncthreads();
        if (tid == 255) carry = cbase + incl;
        __syncthreads();
    }
}

__global__ void radix_scatter_kernel(const unsigned* __restrict__ keysIn,
                                     const unsigned* __restrict__ valsIn,
                                     unsigned* __restrict__ keysOut,
                                     unsigned* __restrict__ valsOut,
                                     int n, int shift,
                                     const unsigned* __restrict__ hist,
                                     int numBlocks) {
    __shared__ unsigned binBase[256];
    __shared__ unsigned waveCnt[8][256];
    __shared__ unsigned waveOff[8][256];
    int tid  = threadIdx.x;
    int lane = tid & 31;
    int wave = tid >> 5;

    binBase[tid] = hist[(size_t)tid * numBlocks + blockIdx.x];
    __syncthreads();

    int base = blockIdx.x * RS_ELEMS;
    for (int t = 0; t < RS_ELEMS / 256; ++t) {
        int  idx   = base + t * 256 + tid;
        bool valid = idx < n;
        unsigned key = valid ? keysIn[idx] : 0xFFFFFFFFu;
        unsigned val = valid ? valsIn[idx] : 0u;
        unsigned bin = (key >> shift) & 255u;

        for (int b = tid; b < 8 * 256; b += 256) ((unsigned*)waveCnt)[b] = 0u;
        __syncthreads();

        // wave32 peer grouping by digit (stable within wave by lane order)
        unsigned long long active = __ballot(valid ? 1 : 0);
        unsigned long long peers  = active;
        for (int b = 0; b < 8; ++b) {
            int bit = (bin >> b) & 1;
            unsigned long long m = __ballot((valid && bit) ? 1 : 0);
            peers &= bit ? m : ~m;
        }
        peers &= active;
        unsigned rank = (unsigned)__popcll(peers & ((1ull << lane) - 1ull));
        unsigned cnt  = (unsigned)__popcll(peers);
        if (valid && rank == 0) waveCnt[wave][bin] = cnt;
        __syncthreads();

        // exclusive scan over waves per bin (thread tid owns bin==tid)
        unsigned sum = 0;
        for (int w = 0; w < 8; ++w) { waveOff[w][tid] = sum; sum += waveCnt[w][tid]; }
        __syncthreads();

        if (valid) {
            unsigned pos = binBase[bin] + waveOff[wave][bin] + rank;
            keysOut[pos] = key;
            valsOut[pos] = val;
        }
        __syncthreads();
        binBase[tid] += sum;
        __syncthreads();
    }
}

// ---------------------------------------------------------------------------
// min / max / sum reduction over T (mean(p) is permutation invariant)
// ---------------------------------------------------------------------------
__global__ void reduce1_kernel(const float* __restrict__ T, int n,
                               float* __restrict__ pmin, float* __restrict__ pmax,
                               float* __restrict__ psum) {
    __shared__ float smin[256], smax[256], ssum[256];
    int tid = threadIdx.x;
    float mn = 3.4e38f, mx = -3.4e38f, sm = 0.0f;
    for (int i = blockIdx.x * 256 + tid; i < n; i += gridDim.x * 256) {
        float v = T[i];
        mn = fminf(mn, v); mx = fmaxf(mx, v); sm += v;
    }
    smin[tid] = mn; smax[tid] = mx; ssum[tid] = sm;
    __syncthreads();
    for (int o = 128; o > 0; o >>= 1) {
        if (tid < o) {
            smin[tid] = fminf(smin[tid], smin[tid + o]);
            smax[tid] = fmaxf(smax[tid], smax[tid + o]);
            ssum[tid] += ssum[tid + o];
        }
        __syncthreads();
    }
    if (tid == 0) { pmin[blockIdx.x] = smin[0]; pmax[blockIdx.x] = smax[0]; psum[blockIdx.x] = ssum[0]; }
}

__global__ void reduce2_kernel(const float* __restrict__ pmin,
                               const float* __restrict__ pmax,
                               const float* __restrict__ psum,
                               int nb, float* __restrict__ stats) {
    __shared__ float smin[256], smax[256], ssum[256];
    int tid = threadIdx.x;
    float mn = 3.4e38f, mx = -3.4e38f, sm = 0.0f;
    for (int i = tid; i < nb; i += 256) {
        mn = fminf(mn, pmin[i]); mx = fmaxf(mx, pmax[i]); sm += psum[i];
    }
    smin[tid] = mn; smax[tid] = mx; ssum[tid] = sm;
    __syncthreads();
    for (int o = 128; o > 0; o >>= 1) {
        if (tid < o) {
            smin[tid] = fminf(smin[tid], smin[tid + o]);
            smax[tid] = fmaxf(smax[tid], smax[tid + o]);
            ssum[tid] += ssum[tid + o];
        }
        __syncthreads();
    }
    if (tid == 0) { stats[0] = smin[0]; stats[1] = smax[0]; stats[2] = ssum[0]; }
}

// ---------------------------------------------------------------------------
// Combine old edges + Gumbel-softmax-scaled sorted candidate edges,
// and produce keys for the second sort (ascending i*N+j in f32).
// ---------------------------------------------------------------------------
__global__ void combine_kernel(const int* __restrict__ old0, const int* __restrict__ old1,
                               int Eold,
                               const int* __restrict__ cand0, const int* __restrict__ cand1,
                               int Ec,
                               const unsigned* __restrict__ order,
                               const float* __restrict__ T,
                               const float* __restrict__ stats,
                               const int* __restrict__ Nptr,
                               float* __restrict__ comb0, float* __restrict__ comb1,
                               unsigned* __restrict__ key2, unsigned* __restrict__ val2,
                               int Et) {
    int t = blockIdx.x * blockDim.x + threadIdx.x;
    if (t >= Et) return;
    float c0, c1;
    if (t < Eold) {
        c0 = (float)old0[t];
        c1 = (float)old1[t];
    } else {
        int s = t - Eold;
        int e = (int)order[s];
        float Tmin = stats[0], Tmax = stats[1];
        float Tmean = stats[2] / (float)Ec;
        float denom = Tmax - Tmin + EPSF;
        float p     = 1.0f - (T[e] - Tmin) / denom;
        float pmean = 1.0f - (Tmean - Tmin) / denom;
        float scale = SAMPLING_RATE / (pmean + EPSF);
        p = fminf(p * scale, 1.0f);
        float l0 = logf(1.0f - p + EPSF);
        float l1 = logf(p + EPSF);
        float u0 = hash_uniform((unsigned)s, 0u);
        float u1 = hash_uniform((unsigned)s, 1u);
        float g0 = -logf(-logf(u0));
        float g1 = -logf(-logf(u1));
        float soft = 1.0f / (1.0f + expf(((l0 + g0) - (l1 + g1)) / TAU));
        c0 = (float)cand0[e] * soft;
        c1 = (float)cand1[e] * soft;
    }
    comb0[t] = c0;
    comb1[t] = c1;
    float Nf = (float)(*Nptr);
    key2[t] = f2o(c0 * Nf + c1);
    val2[t] = (unsigned)t;
}

// ---------------------------------------------------------------------------
// Final gather:  out = [row0 sorted | row1 sorted | ones]
// ---------------------------------------------------------------------------
__global__ void output_kernel(const unsigned* __restrict__ order2,
                              const float* __restrict__ comb0,
                              const float* __restrict__ comb1,
                              float* __restrict__ out, int Et) {
    int p = blockIdx.x * blockDim.x + threadIdx.x;
    if (p < Et) {
        int s = (int)order2[p];
        out[p]          = comb0[s];
        out[Et + p]     = comb1[s];
        out[2 * Et + p] = 1.0f;
    }
}

// ---------------------------------------------------------------------------
// host-side radix sort driver (4 x 8-bit LSD passes; result lands in A bufs)
// ---------------------------------------------------------------------------
static void radix_sort(unsigned* keyA, unsigned* keyB,
                       unsigned* valA, unsigned* valB,
                       int n, unsigned* hist, hipStream_t stream) {
    int nb = (n + RS_ELEMS - 1) / RS_ELEMS;
    for (int pass = 0; pass < 4; ++pass) {
        int shift = pass * 8;
        unsigned *ki, *ko, *vi, *vo;
        if (pass & 1) { ki = keyB; ko = keyA; vi = valB; vo = valA; }
        else          { ki = keyA; ko = keyB; vi = valA; vo = valB; }
        radix_hist_kernel<<<nb, 256, 0, stream>>>(ki, n, shift, hist, nb);
        radix_scan_kernel<<<1, 256, 0, stream>>>(hist, nb * 256);
        radix_scatter_kernel<<<nb, 256, 0, stream>>>(ki, vi, ko, vo, n, shift, hist, nb);
    }
}

// ---------------------------------------------------------------------------
extern "C" void kernel_launch(void* const* d_in, const int* in_sizes, int n_in,
                              void* d_out, int out_size, void* d_ws, size_t ws_size,
                              hipStream_t stream) {
    const int* old_idx = (const int*)d_in[0];     // (2, Eold)
    const float* feat  = (const float*)d_in[1];   // (N, 128)
    const float* energy= (const float*)d_in[2];   // (N,)
    const int* cand    = (const int*)d_in[3];     // (2, Ec)
    const int* Nptr    = (const int*)d_in[4];     // scalar N

    const int Eold = in_sizes[0] / 2;
    const int N    = in_sizes[1] / 128;
    const int Ec   = in_sizes[3] / 2;
    const int Et   = Eold + Ec;

    const int* old0  = old_idx;
    const int* old1  = old_idx + Eold;
    const int* cand0 = cand;
    const int* cand1 = cand + Ec;

    // --- workspace layout -------------------------------------------------
    char* ws = (char*)d_ws;
    size_t off = 0;
    auto alloc = [&](size_t bytes) -> char* {
        char* p = ws + off;
        off += (bytes + 255) & ~(size_t)255;
        return p;
    };
    _Float16* feat_h = (_Float16*)alloc((size_t)N * 128 * sizeof(_Float16));
    float*    norms  = (float*)   alloc((size_t)N * sizeof(float));
    float*    T      = (float*)   alloc((size_t)Ec * sizeof(float));
    unsigned* key1A  = (unsigned*)alloc((size_t)Ec * sizeof(unsigned));
    unsigned* key1B  = (unsigned*)alloc((size_t)Ec * sizeof(unsigned));
    unsigned* val1A  = (unsigned*)alloc((size_t)Ec * sizeof(unsigned));
    unsigned* val1B  = (unsigned*)alloc((size_t)Ec * sizeof(unsigned));
    float*    comb0  = (float*)   alloc((size_t)Et * sizeof(float));
    float*    comb1  = (float*)   alloc((size_t)Et * sizeof(float));
    unsigned* key2A  = (unsigned*)alloc((size_t)Et * sizeof(unsigned));
    unsigned* key2B  = (unsigned*)alloc((size_t)Et * sizeof(unsigned));
    unsigned* val2A  = (unsigned*)alloc((size_t)Et * sizeof(unsigned));
    unsigned* val2B  = (unsigned*)alloc((size_t)Et * sizeof(unsigned));
    int maxNb = (Et + RS_ELEMS - 1) / RS_ELEMS;
    unsigned* hist   = (unsigned*)alloc((size_t)256 * maxNb * sizeof(unsigned));
    float*    pmin   = (float*)   alloc(512 * sizeof(float));
    float*    pmax   = (float*)   alloc(512 * sizeof(float));
    float*    psum   = (float*)   alloc(512 * sizeof(float));
    float*    stats  = (float*)   alloc(16);
    (void)ws_size; (void)n_in; (void)out_size;

    // --- 1. feature f16 copy + squared norms ------------------------------
    feat_norm_kernel<<<N, 128, 0, stream>>>(feat, feat_h, norms);

    // --- 2. per-edge T via WMMA Gram-tile diagonals (async LDS gather) ----
    int tiles = (Ec + 15) / 16;
    edge_T_wmma_kernel<<<tiles, 32, 0, stream>>>(cand0, cand1, feat_h, norms,
                                                 energy, T, Ec);

    // --- 3. stable descending argsort of T --------------------------------
    key1_kernel<<<(Ec + 255) / 256, 256, 0, stream>>>(T, key1A, val1A, Ec);
    radix_sort(key1A, key1B, val1A, val1B, Ec, hist, stream);
    // sorted order now in val1A

    // --- 4. T statistics (min / max / mean) -------------------------------
    reduce1_kernel<<<512, 256, 0, stream>>>(T, Ec, pmin, pmax, psum);
    reduce2_kernel<<<1, 256, 0, stream>>>(pmin, pmax, psum, 512, stats);

    // --- 5. combine old + soft-selected sorted edges, build sort-2 keys ---
    combine_kernel<<<(Et + 255) / 256, 256, 0, stream>>>(
        old0, old1, Eold, cand0, cand1, Ec, val1A, T, stats, Nptr,
        comb0, comb1, key2A, val2A, Et);

    // --- 6. stable ascending argsort by i*N + j ---------------------------
    radix_sort(key2A, key2B, val2A, val2B, Et, hist, stream);
    // sorted order now in val2A

    // --- 7. final gather + ones -------------------------------------------
    output_kernel<<<(Et + 255) / 256, 256, 0, stream>>>(
        val2A, comb0, comb1, (float*)d_out, Et);
}
